// GloveModel_36515811950881
// MI455X (gfx1250) — matrix operand order repaired and checked by reading, not compiled, not run on hardware
//
#include <hip/hip_runtime.h>

typedef __attribute__((ext_vector_type(16))) __bf16        v16bf;
typedef __attribute__((ext_vector_type(8)))  float         v8f;
typedef __attribute__((ext_vector_type(4)))  unsigned int  u32x4;

#define DMODEL 300
#define DPAD   320   // padded K and N (multiple of 64)
#define BATCH  4096

// ---------- helpers ----------

__device__ __forceinline__ unsigned short f2bf(float f) {
  unsigned int u = __float_as_uint(f);
  u += 0x7FFFu + ((u >> 16) & 1u);        // round-to-nearest-even
  return (unsigned short)(u >> 16);
}

__device__ __forceinline__ float waveReduceSum(float v) {
#pragma unroll
  for (int off = 16; off > 0; off >>= 1) v += __shfl_xor(v, off, 32);
  return v;
}

// Packed B layout: element (k,n) lives at Bp[((k>>4)*DPAD + n)*16 + (k&15)].
// A lane's 16 consecutive-K values for one column are 32 contiguous bytes.
__device__ __forceinline__ int bpIdx(int k, int n) {
  return ((k >> 4) * DPAD + n) * 16 + (k & 15);
}

// ---------- M[k][n] = dot(T[k][n][:], tv)  (bf16, packed-B layout) ----------

__global__ __launch_bounds__(256) void k_make_M(const float* __restrict__ T,
                                                const float* __restrict__ tv,
                                                unsigned short* __restrict__ Bp) {
  int wave = threadIdx.x >> 5, lane = threadIdx.x & 31;
  int ei = blockIdx.x * 8 + wave;                 // [0, DPAD*DPAD)
  if (ei >= DPAD * DPAD) return;
  int k = ei / DPAD, n = ei - k * DPAD;
  float s = 0.f;
  if (k < DMODEL && n < DMODEL) {
    const float* row = T + ((size_t)k * DMODEL + n) * DMODEL;
    for (int z = lane; z < DMODEL; z += 32) s += row[z] * tv[z];  // coalesced in z
  }
  s = waveReduceSum(s);
  if (lane == 0)
    Bp[bpIdx(k, n)] = (k < DMODEL && n < DMODEL) ? f2bf(s) : (unsigned short)0;
}

// ---------- Wt[k][n] = Wop[n][k]  (bf16, packed-B layout) ----------

__global__ __launch_bounds__(256) void k_make_Wt(const float* __restrict__ Wop,
                                                 unsigned short* __restrict__ Bp) {
  int id = blockIdx.x * 256 + threadIdx.x;
  if (id >= DPAD * DPAD) return;
  int k = id / DPAD, n = id - k * DPAD;
  unsigned short v = 0;
  if (k < DMODEL && n < DMODEL) v = f2bf(Wop[(size_t)n * DMODEL + k]);
  Bp[bpIdx(k, n)] = v;
}

// ---------- e[b][k] = bf16(embed[idx[b]][k]), zero-padded to DPAD ----------

__global__ __launch_bounds__(256) void k_gather(const float* __restrict__ embed,
                                                const int* __restrict__ indices,
                                                int which,
                                                unsigned short* __restrict__ Ebf) {
  int id = blockIdx.x * 256 + threadIdx.x;        // b*DPAD + k
  int b = id / DPAD, k = id - b * DPAD;
  if (b >= BATCH) return;
  int row = indices[b * 2 + which];
  Ebf[id] = (k < DMODEL) ? f2bf(embed[(size_t)row * DMODEL + k]) : (unsigned short)0;
}

// ---------- bmv[i] = dot(Bm[i][:], tv) ----------

__global__ __launch_bounds__(256) void k_bmv(const float* __restrict__ Bm,
                                             const float* __restrict__ tv,
                                             float* __restrict__ bmv) {
  int wave = threadIdx.x >> 5, lane = threadIdx.x & 31;
  int i = blockIdx.x * 8 + wave;
  if (i >= DMODEL) return;
  float s = 0.f;
  const float* row = Bm + (size_t)i * DMODEL;
  for (int j = lane; j < DMODEL; j += 32) s += row[j] * tv[j];
  s = waveReduceSum(s);
  if (lane == 0) bmv[i] = s;
}

// ---------- WMMA GEMM: C[B][DPAD] = A[B][DPAD] x Bp + biasCol ----------
// Wave tile 16x64 (A fragment reused across 4 N-tiles). 8 waves/WG -> 128x64.

__global__ __launch_bounds__(256)
void k_gemm_wmma(const unsigned short* __restrict__ A,    // bf16 [BATCH][DPAD]
                 const unsigned short* __restrict__ Bp,   // bf16 packed [DPAD][DPAD]
                 const float* __restrict__ biasCol, int biasN,
                 unsigned short* __restrict__ outBf,      // bf16 out (or null)
                 float* __restrict__ outF) {              // f32 out  (or null)
  const int lane = threadIdx.x & 31;
  const int wave = threadIdx.x >> 5;
  const int half = lane >> 4;        // lane-half per ISA fragment layouts
  const int lm   = lane & 15;
  const int rowBase = blockIdx.x * 128 + wave * 16;
  const int colBase = blockIdx.y * 64;

  const v8f vzero = {0.f, 0.f, 0.f, 0.f, 0.f, 0.f, 0.f, 0.f};
  v8f acc[4] = {vzero, vzero, vzero, vzero};

  // A row for this lane (lanes 0-15 and 16-31 both map to rows 0-15 of the tile)
  const u32x4* arow = reinterpret_cast<const u32x4*>(A + (size_t)(rowBase + lm) * DPAD);

  for (int kc = 0; kc < DPAD / 32; ++kc) {
    // A fragment (16x32 bf16): elems 0-15 = K kbase+half*16..
    union { v16bf v; u32x4 u[2]; } af;
    af.u[0] = arow[kc * 4 + 2 * half];
    af.u[1] = arow[kc * 4 + 2 * half + 1];

    const int kb = kc * 2 + half;    // 16-row K-block index for this lane half
    const u32x4* bcol =
        reinterpret_cast<const u32x4*>(Bp) + ((size_t)kb * DPAD + colBase + lm) * 2;
#pragma unroll
    for (int t = 0; t < 4; ++t) {
      union { v16bf v; u32x4 u[2]; } bfm;
      bfm.u[0] = bcol[t * 32 + 0];   // 16 columns * 2 u32x4 per column
      bfm.u[1] = bcol[t * 32 + 1];
      acc[t] = __builtin_amdgcn_wmma_f32_16x16x32_bf16(
          false, af.v, false, bfm.v, (short)0, acc[t], false, false);
    }
  }

#pragma unroll
  for (int t = 0; t < 4; ++t) {
    int col = colBase + t * 16 + lm;
    float bias = (col < biasN) ? biasCol[col] : 0.f;
#pragma unroll
    for (int v = 0; v < 8; ++v) {
      int r = rowBase + half * 8 + v;          // C/D layout: row = v + 8*half
      float val = acc[t][v] + bias;
      if (outBf) outBf[(size_t)r * DPAD + col] = f2bf(val);
      else       outF [(size_t)r * DPAD + col] = val;
    }
  }
}

// ---------- out[b] = dot(hc[b], hx[b]) + b_center[ci] + b_contex[xi] ----------

__global__ __launch_bounds__(256) void k_final(const float* __restrict__ hc,
                                               const float* __restrict__ hx,
                                               const int* __restrict__ indices,
                                               const float* __restrict__ b_center,
                                               const float* __restrict__ b_contex,
                                               float* __restrict__ out) {
  int wave = threadIdx.x >> 5, lane = threadIdx.x & 31;
  int b = blockIdx.x * 8 + wave;
  if (b >= BATCH) return;
  const float* pc = hc + (size_t)b * DPAD;
  const float* px = hx + (size_t)b * DPAD;
  float s = 0.f;
#pragma unroll
  for (int t = 0; t < DPAD / 32; ++t) {    // padded cols are exact zeros
    int d = lane + t * 32;
    s += pc[d] * px[d];
  }
  s = waveReduceSum(s);
  if (lane == 0)
    out[b] = s + b_center[indices[b * 2 + 0]] + b_contex[indices[b * 2 + 1]];
}

// ---------- host ----------

extern "C" void kernel_launch(void* const* d_in, const int* in_sizes, int n_in,
                              void* d_out, int out_size, void* d_ws, size_t ws_size,
                              hipStream_t stream) {
  const int*   indices  = (const int*)d_in[0];
  const float* embed    = (const float*)d_in[1];
  const float* T[2]     = {(const float*)d_in[2],  (const float*)d_in[7]};
  const float* Bm[2]    = {(const float*)d_in[3],  (const float*)d_in[8]};
  const float* tv[2]    = {(const float*)d_in[4],  (const float*)d_in[9]};
  const float* Wop[2]   = {(const float*)d_in[5],  (const float*)d_in[10]};
  const float* bop[2]   = {(const float*)d_in[6],  (const float*)d_in[11]};
  const float* b_center = (const float*)d_in[12];
  const float* b_contex = (const float*)d_in[13];
  float* out = (float*)d_out;

  char* ws = (char*)d_ws;
  size_t off = 0;
  auto wsAlloc = [&](size_t bytes) -> char* {
    char* p = ws + off;
    off = (off + bytes + 255) & ~(size_t)255;
    return p;
  };
  unsigned short* Mb  = (unsigned short*)wsAlloc((size_t)DPAD * DPAD * 2);
  unsigned short* Wt  = (unsigned short*)wsAlloc((size_t)DPAD * DPAD * 2);
  unsigned short* Ebf = (unsigned short*)wsAlloc((size_t)BATCH * DPAD * 2);
  unsigned short* H3  = (unsigned short*)wsAlloc((size_t)BATCH * DPAD * 2);
  float*          bmv = (float*)wsAlloc((size_t)DMODEL * 4);
  float*          Hout[2];
  Hout[0] = (float*)wsAlloc((size_t)BATCH * DPAD * 4);
  Hout[1] = (float*)wsAlloc((size_t)BATCH * DPAD * 4);

  dim3 blk(256);
  dim3 gemmGrid(BATCH / 128, DPAD / 64);
  for (int br = 0; br < 2; ++br) {
    k_bmv    <<<(DMODEL + 7) / 8,    blk, 0, stream>>>(Bm[br], tv[br], bmv);
    k_make_M <<<(DPAD * DPAD) / 8,   blk, 0, stream>>>(T[br], tv[br], Mb);
    k_make_Wt<<<(DPAD * DPAD) / 256, blk, 0, stream>>>(Wop[br], Wt);
    k_gather <<<(BATCH * DPAD) / 256,blk, 0, stream>>>(embed, indices, br, Ebf);
    // h3 = e @ M + Bm@tv  (bf16 out, zero pads preserved)
    k_gemm_wmma<<<gemmGrid, blk, 0, stream>>>(Ebf, Mb, bmv, DMODEL, H3, nullptr);
    // h  = h3 @ Wop^T + bop  (f32 out)
    k_gemm_wmma<<<gemmGrid, blk, 0, stream>>>(H3, Wt, bop[br], DMODEL, nullptr, Hout[br]);
  }
  k_final<<<BATCH / 8, blk, 0, stream>>>(Hout[0], Hout[1], indices, b_center, b_contex, out);

  (void)in_sizes; (void)n_in; (void)out_size; (void)ws_size;
}
